// SEALFullModel_44367012167686
// MI455X (gfx1250) — compile-verified, matrix operand order checked by reading, not run on hardware
//
#include <hip/hip_runtime.h>

typedef __attribute__((ext_vector_type(16))) _Float16 v16h;
typedef __attribute__((ext_vector_type(8)))  _Float16 v8h;
typedef __attribute__((ext_vector_type(8)))  float    v8f;
typedef __attribute__((ext_vector_type(4)))  unsigned v4u;
typedef __attribute__((ext_vector_type(8)))  int      v8i;
typedef __attribute__((ext_vector_type(4)))  int      v4i;

namespace {
constexpr int kN   = 64;           // nodes per graph
constexpr int kHID = 32;           // GIN hidden dim
constexpr int kLD  = 16;           // embedding dim
constexpr int kDEG = 16;
constexpr int kE   = kN * kDEG;    // 1024 edges per graph
constexpr int kK   = 10;           // SortPool k
constexpr int kCH1 = 32;
constexpr int kCH2 = 16;
}

union V16H { v16h v; v8h h[2]; _Float16 e[16]; };
union V8F  { v8f  v; float e[8]; };

// ---------------- Tensor Data Mover (gfx1250) ----------------
#ifndef USE_TDM
#define USE_TDM 1
#endif
#if USE_TDM && __has_builtin(__builtin_amdgcn_tensor_load_to_lds) && \
    __has_builtin(__builtin_amdgcn_s_wait_tensorcnt)
#define TDM_OK 1
#else
#define TDM_OK 0
#endif

#if TDM_OK
__device__ inline unsigned lds_addr_of(const void* p) {
  // generic -> LDS addrspace pointer; its integer value is the LDS byte offset
  return (unsigned)(unsigned long long)(const __attribute__((address_space(3))) char*)p;
}

// 1-D contiguous DWORD-tensor load: nelem u32 words from gsrc into LDS.
__device__ inline void tdm_load_1d(const void* gsrc, unsigned lds_off, int nelem) {
  const unsigned long long ga = (unsigned long long)gsrc;
  v4u g0;
  g0.x = 1u;                                        // count=1 (valid user D#)
  g0.y = lds_off;                                   // lds_addr
  g0.z = (unsigned)(ga & 0xffffffffu);              // global_addr[31:0]
  g0.w = (unsigned)((ga >> 32) & 0x01ffffffu)       // global_addr[56:32]
         | (2u << 30);                              // type = 2 ("image")
  v8i g1;
  g1[0] = (int)(2u << 16);                          // wg_mask=0, data_size=2 (4B)
  g1[1] = (int)(((unsigned)nelem & 0xffffu) << 16); // tensor_dim0[15:0] @ bits 63:48
  g1[2] = (int)((((unsigned)nelem >> 16) & 0xffffu) // tensor_dim0[31:16]
                | (1u << 16));                      // tensor_dim1 = 1
  g1[3] = (int)(((unsigned)nelem & 0xffffu) << 16); // tile_dim0 @ bits 127:112
  g1[4] = 1;                                        // tile_dim1 = 1
  g1[5] = nelem;                                    // tensor_dim0_stride
  g1[6] = 0;
  g1[7] = 0;
  v4i z4 = {0, 0, 0, 0};
#if defined(__clang_major__) && (__clang_major__ >= 23)
  v8i z8 = {0, 0, 0, 0, 0, 0, 0, 0};
  __builtin_amdgcn_tensor_load_to_lds(g0, g1, z4, z4, z8, 0);
#else
  __builtin_amdgcn_tensor_load_to_lds(g0, g1, z4, z4, 0);
#endif
}
#endif  // TDM_OK

__global__ __launch_bounds__(256)
void seal_full_kernel(
    const int* __restrict__ labels, const int* __restrict__ src,
    const int* __restrict__ dst,    const float* __restrict__ emb,
    const float* __restrict__ w1_0, const float* __restrict__ b1_0,
    const float* __restrict__ w2_0, const float* __restrict__ b2_0,
    const float* __restrict__ w1_1, const float* __restrict__ b1_1,
    const float* __restrict__ w2_1, const float* __restrict__ b2_1,
    const float* __restrict__ w1_2, const float* __restrict__ b1_2,
    const float* __restrict__ w2_2, const float* __restrict__ b2_2,
    const float* __restrict__ conv1_w, const float* __restrict__ conv1_b,
    const float* __restrict__ conv2_w, const float* __restrict__ conv2_b,
    const float* __restrict__ sc_w1,   const float* __restrict__ sc_b1,
    const float* __restrict__ sc_w2,   const float* __restrict__ sc_b2,
    float* __restrict__ out)
{
  __shared__ alignas(16) float    sh_h[kN][kHID];       // node features (f32)
  __shared__ alignas(16) float    sh_agg[kN][kHID];     // neighbor aggregation
  __shared__ int      sh_srcraw[kE];
  __shared__ int      sh_dstraw[kE];
  __shared__ int      sh_edge[kE];                      // packed local src|dst
  __shared__ alignas(16) _Float16 sh_za[kN][kHID];      // WMMA A (z = h+agg)
  __shared__ alignas(16) _Float16 sh_ta[kN][kHID];      // WMMA A (t)
  __shared__ alignas(16) _Float16 sh_wT[6][kHID][kHID]; // weights, f16, TRANSPOSED [n][k]
  __shared__ float    sh_bias[6][kHID];
  __shared__ int      sh_lab[kN];
  __shared__ float    sh_poolp[kK + 4][kHID];           // pad-2 halo rows
  __shared__ float    sh_c1[kCH1][kK];
  __shared__ float    sh_p1p[kCH1][kK / 2 + 4];         // pad-2 halo cols
  __shared__ float    sh_c2[kCH2][kK / 2];
  __shared__ float    sh_mean[kCH2];
  __shared__ float    sh_sv[kCH2];

  const int g    = blockIdx.x;
  const int tid  = threadIdx.x;
  const int lane = tid & 31;
  const int wave = tid >> 5;          // 8 waves

  const int ebase = g * kE;

  // ---- kick off async edge-block DMA (TDM), overlap with embedding gather ----
#if TDM_OK
  if (wave == 0) {
    tdm_load_1d(src + ebase, lds_addr_of(&sh_srcraw[0]), kE);
    tdm_load_1d(dst + ebase, lds_addr_of(&sh_dstraw[0]), kE);
  }
#else
  for (int e = tid; e < kE; e += 256) {
    sh_srcraw[e] = src[ebase + e];
    sh_dstraw[e] = dst[ebase + e];
  }
#endif

  // warm L2/WGP$ for the conv weights used at the tail (global_prefetch_b8)
  __builtin_prefetch(conv1_w + tid, 0, 1);
  __builtin_prefetch(conv2_w + tid, 0, 1);

  // ---- embedding gather (zero-padded LD=16 -> 32 columns) ----
  if (tid < kN) sh_lab[tid] = labels[g * kN + tid];
  __syncthreads();
  for (int idx = tid; idx < kN * kHID; idx += 256) {
    const int n = idx >> 5, c = idx & 31;
    sh_h[n][c] = (c < kLD) ? emb[sh_lab[n] * kLD + c] : 0.0f;
  }

  // ---- stage all 6 GIN weight matrices once: f16, transposed [n][k] ----
  {
    const float* Wp[6]   = { w1_0, w2_0, w1_1, w2_1, w1_2, w2_2 };
    const float* Bp[6]   = { b1_0, b2_0, b1_1, b2_1, b1_2, b2_2 };
    const int    kdim[6] = { kLD, kHID, kHID, kHID, kHID, kHID };
    for (int idx = tid; idx < 6 * kHID * kHID; idx += 256) {
      const int m = idx >> 10;
      const int k = (idx >> 5) & 31;    // contiguous n for coalesced global read
      const int n = idx & 31;
      sh_wT[m][n][k] = (_Float16)((k < kdim[m]) ? Wp[m][k * kHID + n] : 0.0f);
    }
    for (int idx = tid; idx < 6 * kHID; idx += 256)
      sh_bias[idx >> 5][idx & 31] = Bp[idx >> 5][idx & 31];
  }

#if TDM_OK
  if (wave == 0) __builtin_amdgcn_s_wait_tensorcnt(0);
#endif
  __syncthreads();

  // ---- localize edges: pack (src-local | dst-local<<16) ----
  for (int e = tid; e < kE; e += 256)
    sh_edge[e] = ((sh_srcraw[e] - g * kN) & 0xffff) | ((sh_dstraw[e] - g * kN) << 16);

  // WMMA tile assignment: wave w -> (M-tile, N-tile); 4x2 tiles cover 64x32
  const int mt   = wave >> 1;               // 0..3
  const int nt   = wave & 1;                // 0..1
  const int rowA = (lane & 15) + mt * 16;   // A-matrix row for this lane
  const int colB = (lane & 15) + nt * 16;   // B/D column for this lane
  const int kgrp = (lane >> 4) * 8;         // K-group offset (lanes 16-31)
  const int rhi  = (lane >> 4) * 8;         // D-fragment row offset

  for (int l = 0; l < 3; ++l) {
    // zero aggregation buffer
    for (int idx = tid; idx < kN * kHID; idx += 256)
      (&sh_agg[0][0])[idx] = 0.0f;
    __syncthreads();

    // scatter-add: agg[dst] += h[src]  (ds_load_b128 + 4x ds_add_f32)
    for (int idx = tid; idx < kE * 8; idx += 256) {
      const int e  = idx >> 3;
      const int c0 = (idx & 7) * 4;
      const int pk = sh_edge[e];
      const int sl = pk & 0xffff, dl = pk >> 16;
      const float4 hv = *(const float4*)&sh_h[sl][c0];
      atomicAdd(&sh_agg[dl][c0 + 0], hv.x);
      atomicAdd(&sh_agg[dl][c0 + 1], hv.y);
      atomicAdd(&sh_agg[dl][c0 + 2], hv.z);
      atomicAdd(&sh_agg[dl][c0 + 3], hv.w);
    }
    __syncthreads();

    // z = h + agg, down-convert to f16 A-operand
    for (int idx = tid; idx < kN * kHID; idx += 256) {
      const int n = idx >> 5, c = idx & 31;
      sh_za[n][c] = (_Float16)(sh_h[n][c] + sh_agg[n][c]);
    }
    __syncthreads();

    // ---- WMMA 1: t = relu(z @ W1 + b1); fragments = 2x ds_load_b128 each ----
    {
      V16H a, b; V8F c = {};
      a.h[0] = *(const v8h*)&sh_za[rowA][kgrp];
      a.h[1] = *(const v8h*)&sh_za[rowA][kgrp + 16];
      b.h[0] = *(const v8h*)&sh_wT[2 * l][colB][kgrp];
      b.h[1] = *(const v8h*)&sh_wT[2 * l][colB][kgrp + 16];
      c.v = __builtin_amdgcn_wmma_f32_16x16x32_f16(
          false, a.v, false, b.v, (short)0, c.v, false, false);
      const float bs = sh_bias[2 * l][colB];
#pragma unroll
      for (int r = 0; r < 8; ++r) {
        const int m = mt * 16 + rhi + r;
        sh_ta[m][colB] = (_Float16)fmaxf(c.e[r] + bs, 0.0f);
      }
    }
    __syncthreads();

    // ---- WMMA 2: h = relu(t @ W2 + b2) ----
    {
      V16H a, b; V8F c = {};
      a.h[0] = *(const v8h*)&sh_ta[rowA][kgrp];
      a.h[1] = *(const v8h*)&sh_ta[rowA][kgrp + 16];
      b.h[0] = *(const v8h*)&sh_wT[2 * l + 1][colB][kgrp];
      b.h[1] = *(const v8h*)&sh_wT[2 * l + 1][colB][kgrp + 16];
      c.v = __builtin_amdgcn_wmma_f32_16x16x32_f16(
          false, a.v, false, b.v, (short)0, c.v, false, false);
      const float bs = sh_bias[2 * l + 1][colB];
#pragma unroll
      for (int r = 0; r < 8; ++r) {
        const int m = mt * 16 + rhi + r;
        sh_h[m][colB] = fmaxf(c.e[r] + bs, 0.0f);
      }
    }
    __syncthreads();
  }

  // ---- zero padded conv buffers (halos stay zero -> branch-free convs) ----
  for (int idx = tid; idx < (kK + 4) * kHID; idx += 256)
    (&sh_poolp[0][0])[idx] = 0.0f;
  for (int idx = tid; idx < kCH1 * (kK / 2 + 4); idx += 256)
    (&sh_p1p[0][0])[idx] = 0.0f;
  __syncthreads();

  // ---- SortPooling: stable rank by channel 31 descending, keep top-10 ----
  if (tid < kN) {
    const float key = sh_h[tid][kHID - 1];
    int rank = 0;
    for (int j = 0; j < kN; ++j) {
      const float kj = sh_h[j][kHID - 1];
      rank += (kj > key) || (kj == key && j < tid);
    }
    if (rank < kK)
      for (int c = 0; c < kHID; ++c) sh_poolp[rank + 2][c] = sh_h[tid][c];
  }
  __syncthreads();

  // ---- Conv1d(32->32, k=5, pad 2) + ReLU (branch-free via halo) ----
  for (int idx = tid; idx < kCH1 * kK; idx += 256) {
    const int o = idx / kK, p = idx % kK;
    float acc = conv1_b[o];
    const float* wrow = conv1_w + o * (kHID * 5);
    for (int i = 0; i < kHID; ++i)
#pragma unroll
      for (int kk = 0; kk < 5; ++kk)
        acc += wrow[i * 5 + kk] * sh_poolp[p + kk][i];
    sh_c1[o][p] = fmaxf(acc, 0.0f);
  }
  __syncthreads();

  // ---- MaxPool1d(2,2): 10 -> 5 (into padded buffer) ----
  for (int idx = tid; idx < kCH1 * 5; idx += 256) {
    const int o = idx / 5, q = idx % 5;
    sh_p1p[o][q + 2] = fmaxf(sh_c1[o][2 * q], sh_c1[o][2 * q + 1]);
  }
  __syncthreads();

  // ---- Conv1d(32->16, k=5, pad 2) + ReLU ----
  for (int idx = tid; idx < kCH2 * 5; idx += 256) {
    const int o = idx / 5, p = idx % 5;
    float acc = conv2_b[o];
    const float* wrow = conv2_w + o * (kCH1 * 5);
    for (int i = 0; i < kCH1; ++i)
#pragma unroll
      for (int kk = 0; kk < 5; ++kk)
        acc += wrow[i * 5 + kk] * sh_p1p[i][p + kk];
    sh_c2[o][p] = fmaxf(acc, 0.0f);
  }
  __syncthreads();

  // ---- AdaptiveAvgPool1d(1) ----
  if (tid < kCH2) {
    float s = 0.0f;
    for (int p = 0; p < 5; ++p) s += sh_c2[tid][p];
    sh_mean[tid] = s * 0.2f;
  }
  __syncthreads();

  // ---- scorer MLP ----
  if (tid < kCH2) {
    float acc = sc_b1[tid];
    for (int o = 0; o < kCH2; ++o) acc += sh_mean[o] * sc_w1[o * kCH2 + tid];
    sh_sv[tid] = fmaxf(acc, 0.0f);
  }
  __syncthreads();
  if (tid == 0) {
    float s = sc_b2[0];
    for (int j = 0; j < kCH2; ++j) s += sh_sv[j] * sc_w2[j];
    out[g] = s;
  }
}

extern "C" void kernel_launch(void* const* d_in, const int* in_sizes, int n_in,
                              void* d_out, int out_size, void* d_ws, size_t ws_size,
                              hipStream_t stream) {
  (void)n_in; (void)d_ws; (void)ws_size; (void)out_size;
  const int*   labels  = (const int*)  d_in[0];
  const int*   src     = (const int*)  d_in[1];
  const int*   dst     = (const int*)  d_in[2];
  const float* emb     = (const float*)d_in[3];
  const float* w1_0    = (const float*)d_in[4];
  const float* b1_0    = (const float*)d_in[5];
  const float* w2_0    = (const float*)d_in[6];
  const float* b2_0    = (const float*)d_in[7];
  const float* w1_1    = (const float*)d_in[8];
  const float* b1_1    = (const float*)d_in[9];
  const float* w2_1    = (const float*)d_in[10];
  const float* b2_1    = (const float*)d_in[11];
  const float* w1_2    = (const float*)d_in[12];
  const float* b1_2    = (const float*)d_in[13];
  const float* w2_2    = (const float*)d_in[14];
  const float* b2_2    = (const float*)d_in[15];
  const float* conv1_w = (const float*)d_in[16];
  const float* conv1_b = (const float*)d_in[17];
  const float* conv2_w = (const float*)d_in[18];
  const float* conv2_b = (const float*)d_in[19];
  const float* sc_w1   = (const float*)d_in[20];
  const float* sc_b1   = (const float*)d_in[21];
  const float* sc_w2   = (const float*)d_in[22];
  const float* sc_b2   = (const float*)d_in[23];
  float* out = (float*)d_out;

  const int B = in_sizes[0] / kN;   // 4096 graphs, one workgroup each
  seal_full_kernel<<<B, 256, 0, stream>>>(
      labels, src, dst, emb,
      w1_0, b1_0, w2_0, b2_0,
      w1_1, b1_1, w2_1, b2_1,
      w1_2, b1_2, w2_2, b2_2,
      conv1_w, conv1_b, conv2_w, conv2_b,
      sc_w1, sc_b1, sc_w2, sc_b2, out);
}
